// ICNNBlock_44581760532841
// MI455X (gfx1250) — compile-verified, matrix operand order checked by reading, not run on hardware
//
#include <hip/hip_runtime.h>

// ---------------------------------------------------------------------------
// ICNN 4-layer forward for MI455X (gfx1250, wave32, WMMA bf16).
//   z = relu( xc@Wc^T + xf@Wu^T + z_prev@softplus(U)^T
//             + relu(xf@Wut^T + but) + bc + bu )
// Single accumulator set: u-phase first, folded in place, then main phases.
// Global->LDS staging uses gfx1250 GLOBAL_LOAD_ASYNC_TO_LDS_B128 with an
// explicit double-buffered ASYNCcnt pipeline (sync fallback if builtin absent).
// ---------------------------------------------------------------------------

typedef __attribute__((ext_vector_type(16))) __bf16        v16bf;
typedef __attribute__((ext_vector_type(8)))  float         v8f;
typedef __attribute__((ext_vector_type(4)))  unsigned int  v4u;
typedef __attribute__((ext_vector_type(4)))  int           v4i;

typedef __attribute__((address_space(1))) v4i* gv4i_p;   // global 16B chunk
typedef __attribute__((address_space(3))) v4i* lv4i_p;   // LDS    16B chunk

#define BATCH  8192
#define NIN    256
#define HID    1024
#define NLAYER 4

#define BM 128          // block tile M
#define BN 256          // block tile N
#define BK 32           // K step (one bf16 WMMA depth)
#define STRIDE 40       // halves per LDS row (32 data + 8 pad) -> 80B

#define A_CH 2          // 16B chunks per thread for A tile (128*4 / 256)
#define B_CH 4          // 16B chunks per thread for B tile (256*4 / 256)
#define ISSUES (A_CH + B_CH)

#if defined(__has_builtin)
#if __has_builtin(__builtin_amdgcn_global_load_async_to_lds_b128) && \
    __has_builtin(__builtin_amdgcn_s_wait_asynccnt)
#define HAVE_ASYNC_LDS 1
#else
#define HAVE_ASYNC_LDS 0
#endif
#else
#define HAVE_ASYNC_LDS 0
#endif

__device__ __forceinline__ unsigned short f2bf(float f) {
    unsigned int u = __float_as_uint(f);
    u += 0x7fffu + ((u >> 16) & 1u);          // round-to-nearest-even
    return (unsigned short)(u >> 16);
}

__device__ __forceinline__ float softplusf(float x) {
    return (x > 20.0f) ? x : log1pf(expf(x));
}

// ---------------------------- prep kernels ---------------------------------

__global__ void icnn_prep_x(const float* __restrict__ x,
                            unsigned short* __restrict__ o, int n) {
    for (int i = blockIdx.x * blockDim.x + threadIdx.x; i < n;
         i += gridDim.x * blockDim.x)
        o[i] = f2bf(x[i]);
}

__global__ void icnn_prep_w(const float* __restrict__ w,
                            unsigned short* __restrict__ o, int n) {
    for (int i = blockIdx.x * blockDim.x + threadIdx.x; i < n;
         i += gridDim.x * blockDim.x) {
        int col = i & (NIN - 1);
        float v = w[i];
        if (col < 8) {
            float s = (col < 4) ? 1.0f : -1.0f;
            v = s * softplusf(v);
        }
        o[i] = f2bf(v);
    }
}

__global__ void icnn_prep_u(const float* __restrict__ u,
                            unsigned short* __restrict__ o, int n) {
    for (int i = blockIdx.x * blockDim.x + threadIdx.x; i < n;
         i += gridDim.x * blockDim.x)
        o[i] = f2bf(softplusf(u[i]));
}

// ---------------------------- pipeline helpers -----------------------------

template <int N>
__device__ __forceinline__ void wait_async() {
#if HAVE_ASYNC_LDS
    __builtin_amdgcn_s_wait_asynccnt(N);
#endif
}

// Stage one BK-deep slab of A (128 rows) and B (256 rows) into LDS buffers.
__device__ __forceinline__ void issue_tile(
    const unsigned short* __restrict__ A, int lda,
    const unsigned short* __restrict__ B, int ldb,
    int m0, int n0, int k0,
    unsigned short* Asb, unsigned short* Bsb, int t)
{
#if HAVE_ASYNC_LDS
    #pragma unroll
    for (int i = 0; i < A_CH; ++i) {
        int c = t + 256 * i;
        int row = c >> 2, chn = (c & 3) * 8;
        __builtin_amdgcn_global_load_async_to_lds_b128(
            (gv4i_p)(A + (size_t)(m0 + row) * lda + k0 + chn),
            (lv4i_p)(Asb + row * STRIDE + chn),
            0, 0);
    }
    #pragma unroll
    for (int i = 0; i < B_CH; ++i) {
        int c = t + 256 * i;
        int row = c >> 2, chn = (c & 3) * 8;
        __builtin_amdgcn_global_load_async_to_lds_b128(
            (gv4i_p)(B + (size_t)(n0 + row) * ldb + k0 + chn),
            (lv4i_p)(Bsb + row * STRIDE + chn),
            0, 0);
    }
#else
    #pragma unroll
    for (int i = 0; i < A_CH; ++i) {
        int c = t + 256 * i;
        int row = c >> 2, chn = (c & 3) * 8;
        v4u d = *(const v4u*)(A + (size_t)(m0 + row) * lda + k0 + chn);
        *(v4u*)(Asb + row * STRIDE + chn) = d;
    }
    #pragma unroll
    for (int i = 0; i < B_CH; ++i) {
        int c = t + 256 * i;
        int row = c >> 2, chn = (c & 3) * 8;
        v4u d = *(const v4u*)(B + (size_t)(n0 + row) * ldb + k0 + chn);
        *(v4u*)(Bsb + row * STRIDE + chn) = d;
    }
#endif
}

// One K-step of WMMAs from staged LDS. Wave tile 64x64 = 4x4 fragments.
// 16-bit fragment layout (ISA 7.12.2): lanes 0-15 hold K 0-7 & 16-23,
// lanes 16-31 hold K 8-15 & 24-31; B mirrors A with N across lanes.
__device__ __forceinline__ void compute_step(
    const unsigned short* Asb, const unsigned short* Bsb,
    v8f (&acc)[4][4], int wm, int wn, int lane)
{
    const int lhalf = lane & 15;
    const int hsel  = (lane >> 4) & 1;
    v16bf a[4], b[4];
    #pragma unroll
    for (int mf = 0; mf < 4; ++mf) {
        const unsigned short* ap =
            Asb + (wm * 64 + mf * 16 + lhalf) * STRIDE + hsel * 8;
        union { v4u q[2]; v16bf v; } uu;
        uu.q[0] = *(const v4u*)(ap);
        uu.q[1] = *(const v4u*)(ap + 16);
        a[mf] = uu.v;
    }
    #pragma unroll
    for (int nf = 0; nf < 4; ++nf) {
        const unsigned short* bp =
            Bsb + (wn * 64 + nf * 16 + lhalf) * STRIDE + hsel * 8;
        union { v4u q[2]; v16bf v; } uu;
        uu.q[0] = *(const v4u*)(bp);
        uu.q[1] = *(const v4u*)(bp + 16);
        b[nf] = uu.v;
    }
    #pragma unroll
    for (int mf = 0; mf < 4; ++mf)
        #pragma unroll
        for (int nf = 0; nf < 4; ++nf)
            acc[mf][nf] = __builtin_amdgcn_wmma_f32_16x16x32_bf16(
                false, a[mf], false, b[nf],
                (short)0, acc[mf][nf], false, false);
}

// Full GEMM phase: acc += A[m0.., :K] * B[n0.., :K]^T, double-buffered.
__device__ __forceinline__ void gemm_phase(
    const unsigned short* __restrict__ A, int lda,
    const unsigned short* __restrict__ B, int ldb, int K,
    int m0, int n0,
    unsigned short (&As)[2][BM * STRIDE],
    unsigned short (&Bs)[2][BN * STRIDE],
    v8f (&acc)[4][4], int wm, int wn, int lane)
{
    const int t = threadIdx.x;
    issue_tile(A, lda, B, ldb, m0, n0, 0, As[0], Bs[0], t);
    int buf = 0;
    for (int k0 = 0; k0 + BK < K; k0 += BK) {
        issue_tile(A, lda, B, ldb, m0, n0, k0 + BK, As[buf ^ 1], Bs[buf ^ 1], t);
        wait_async<ISSUES>();     // this buffer's loads (older group) landed
        __syncthreads();          // ...for every wave in the block
        compute_step(As[buf], Bs[buf], acc, wm, wn, lane);
        __syncthreads();          // everyone done reading before overwrite
        buf ^= 1;
    }
    wait_async<0>();
    __syncthreads();
    compute_step(As[buf], Bs[buf], acc, wm, wn, lane);
    __syncthreads();
}

// ---------------------------- fused layer ----------------------------------

template <bool HAS_U, bool LAST>
__global__ __launch_bounds__(256)
void icnn_layer(const unsigned short* __restrict__ xc,
                const unsigned short* __restrict__ xf,
                const unsigned short* __restrict__ zin,
                const unsigned short* __restrict__ Wc,
                const unsigned short* __restrict__ Wu,
                const unsigned short* __restrict__ Wut,
                const unsigned short* __restrict__ U,
                const float* __restrict__ bc,
                const float* __restrict__ bu,
                const float* __restrict__ but,
                unsigned short* __restrict__ zout_bf,
                float* __restrict__ zout_f32)
{
    __shared__ __align__(16) unsigned short As[2][BM * STRIDE];
    __shared__ __align__(16) unsigned short Bs[2][BN * STRIDE];

    const int n0    = blockIdx.x * BN;
    const int m0    = blockIdx.y * BM;
    const int lane  = threadIdx.x & 31;
    const int wave  = threadIdx.x >> 5;
    const int wm    = wave >> 2;          // 0..1  (M)
    const int wn    = wave & 3;           // 0..3  (N)
    const int lhalf = lane & 15;
    const int hsel  = (lane >> 4) & 1;

    v8f acc[4][4];

    // ---- phase 1: u = xf @ Wut^T + but ----
    #pragma unroll
    for (int nf = 0; nf < 4; ++nf) {
        float b2 = but[n0 + wn * 64 + nf * 16 + lhalf];
        #pragma unroll
        for (int mf = 0; mf < 4; ++mf)
            #pragma unroll
            for (int r = 0; r < 8; ++r)
                acc[mf][nf][r] = b2;
    }
    gemm_phase(xf, NIN, Wut, NIN, NIN, m0, n0, As, Bs, acc, wm, wn, lane);

    // fold in place: acc <- (bc + bu) + relu(acc)
    #pragma unroll
    for (int nf = 0; nf < 4; ++nf) {
        int col  = n0 + wn * 64 + nf * 16 + lhalf;
        float bm = bc[col] + bu[col];
        #pragma unroll
        for (int mf = 0; mf < 4; ++mf)
            #pragma unroll
            for (int r = 0; r < 8; ++r)
                acc[mf][nf][r] = bm + fmaxf(acc[mf][nf][r], 0.0f);
    }

    // ---- phase 2: main paths accumulate into same registers ----
    gemm_phase(xc, NIN, Wc, NIN, NIN, m0, n0, As, Bs, acc, wm, wn, lane);
    gemm_phase(xf, NIN, Wu, NIN, NIN, m0, n0, As, Bs, acc, wm, wn, lane);
    if (HAS_U)
        gemm_phase(zin, HID, U, HID, HID, m0, n0, As, Bs, acc, wm, wn, lane);

    // ---- epilogue: z = relu(acc); C/D layout: vgpr r -> M = r + 8*hsel ----
    #pragma unroll
    for (int mf = 0; mf < 4; ++mf)
        #pragma unroll
        for (int nf = 0; nf < 4; ++nf) {
            int col = n0 + wn * 64 + nf * 16 + lhalf;
            #pragma unroll
            for (int r = 0; r < 8; ++r) {
                int row = m0 + wm * 64 + mf * 16 + r + hsel * 8;
                float v = fmaxf(acc[mf][nf][r], 0.0f);
                size_t off = (size_t)row * HID + col;
                if (LAST) zout_f32[off] = v;
                else      zout_bf[off]  = f2bf(v);
            }
        }
}

// ---------------------------- launcher -------------------------------------

extern "C" void kernel_launch(void* const* d_in, const int* in_sizes, int n_in,
                              void* d_out, int out_size, void* d_ws,
                              size_t ws_size, hipStream_t stream)
{
    const float* xc    = (const float*)d_in[0];
    const float* xf    = (const float*)d_in[1];
    const float* Wc_w  = (const float*)d_in[2];
    const float* Wc_b  = (const float*)d_in[3];
    const float* Wu_w  = (const float*)d_in[4];
    const float* Wu_b  = (const float*)d_in[5];
    const float* Wut_w = (const float*)d_in[6];
    const float* Wut_b = (const float*)d_in[7];
    const float* raw_U = (const float*)d_in[8];

    unsigned short* ws = (unsigned short*)d_ws;
    size_t off = 0;
    unsigned short* xcb = ws + off; off += (size_t)BATCH * NIN;
    unsigned short* xfb = ws + off; off += (size_t)BATCH * NIN;
    unsigned short* Wcb = ws + off; off += (size_t)NLAYER * HID * NIN;
    unsigned short* Wub = ws + off; off += (size_t)NLAYER * HID * NIN;
    unsigned short* Wtb = ws + off; off += (size_t)NLAYER * HID * NIN;
    unsigned short* Ub  = ws + off; off += (size_t)(NLAYER - 1) * HID * HID;
    unsigned short* z0  = ws + off; off += (size_t)BATCH * HID;
    unsigned short* z1  = ws + off; off += (size_t)BATCH * HID;

    dim3 pb(256), pg(512);
    icnn_prep_x<<<pg, pb, 0, stream>>>(xc, xcb, BATCH * NIN);
    icnn_prep_x<<<pg, pb, 0, stream>>>(xf, xfb, BATCH * NIN);
    icnn_prep_w<<<pg, pb, 0, stream>>>(Wc_w,  Wcb, NLAYER * HID * NIN);
    icnn_prep_w<<<pg, pb, 0, stream>>>(Wu_w,  Wub, NLAYER * HID * NIN);
    icnn_prep_w<<<pg, pb, 0, stream>>>(Wut_w, Wtb, NLAYER * HID * NIN);
    icnn_prep_u<<<pg, pb, 0, stream>>>(raw_U, Ub, (NLAYER - 1) * HID * HID);

    dim3 blk(256);
    dim3 grd(HID / BN, BATCH / BM);   // 4 x 64 = 256 blocks
    float* out = (float*)d_out;

    icnn_layer<false, false><<<grd, blk, 0, stream>>>(
        xcb, xfb, nullptr,
        Wcb, Wub, Wtb, nullptr,
        Wc_b, Wu_b, Wut_b, z0, nullptr);

    icnn_layer<true, false><<<grd, blk, 0, stream>>>(
        xcb, xfb, z0,
        Wcb + 1 * HID * NIN, Wub + 1 * HID * NIN, Wtb + 1 * HID * NIN,
        Ub + 0 * HID * HID,
        Wc_b + 1 * HID, Wu_b + 1 * HID, Wut_b + 1 * HID, z1, nullptr);

    icnn_layer<true, false><<<grd, blk, 0, stream>>>(
        xcb, xfb, z1,
        Wcb + 2 * HID * NIN, Wub + 2 * HID * NIN, Wtb + 2 * HID * NIN,
        Ub + 1 * HID * HID,
        Wc_b + 2 * HID, Wu_b + 2 * HID, Wut_b + 2 * HID, z0, nullptr);

    icnn_layer<true, true><<<grd, blk, 0, stream>>>(
        xcb, xfb, z0,
        Wcb + 3 * HID * NIN, Wub + 3 * HID * NIN, Wtb + 3 * HID * NIN,
        Ub + 2 * HID * HID,
        Wc_b + 3 * HID, Wu_b + 3 * HID, Wut_b + 3 * HID, nullptr, out);
}
